// contrastive_kd_64372969832695
// MI455X (gfx1250) — matrix operand order repaired and checked by reading, not compiled, not run on hardware
//
#include <hip/hip_runtime.h>
#include <hip/hip_bf16.h>

typedef _Float16 v2h  __attribute__((ext_vector_type(2)));
typedef _Float16 v8h  __attribute__((ext_vector_type(8)));
typedef _Float16 v16h __attribute__((ext_vector_type(16)));
typedef float    v8f  __attribute__((ext_vector_type(8)));

#define N_TOTAL 8192
#define DIM     64
#define INV_T   10.0f
#define LOG2E   1.44269504088896340736f
#define LN2     0.69314718055994530942f
#define EPS_N   1e-8f
#define NCHUNK  16              // column chunks per pass
#define TILES   (512 / NCHUNK)  // 16-wide column tiles per block (32)
#define SUBT    16              // tiles staged per LDS phase (256 cols = 36 KB)
#define NSTAGE  (TILES / SUBT)  // 2 staging phases

// ---------------- wave32 reductions ----------------
__device__ __forceinline__ float wave_sum32(float v) {
    v += __shfl_xor(v, 1, 32);
    v += __shfl_xor(v, 2, 32);
    v += __shfl_xor(v, 4, 32);
    v += __shfl_xor(v, 8, 32);
    v += __shfl_xor(v, 16, 32);
    return v;
}
// sum within each 16-lane half of a wave32 (xor masks < 16 stay in-group)
__device__ __forceinline__ float half_sum16(float v) {
    v += __shfl_xor(v, 1, 32);
    v += __shfl_xor(v, 2, 32);
    v += __shfl_xor(v, 4, 32);
    v += __shfl_xor(v, 8, 32);
    return v;
}

// ---------------- zero the row/col accumulators ----------------
__global__ void zero_kernel(float* __restrict__ p, int n) {
    int i = blockIdx.x * blockDim.x + threadIdx.x;
    if (i < n) p[i] = 0.0f;
}

// ---------------- normalize fp32 -> unit f16, plus scaled diagonal ----------------
// one wave32 per row index i: handles ts[i] and img[i] (64 floats each, 2/lane)
__global__ void normalize_kernel(const float* __restrict__ ts,
                                 const float* __restrict__ img,
                                 _Float16* __restrict__ a_h,
                                 _Float16* __restrict__ b_h,
                                 float* __restrict__ diag) {
    const int lane = threadIdx.x & 31;
    const int row  = (blockIdx.x * blockDim.x + threadIdx.x) >> 5;

    const float2 va = ((const float2*)(ts  + row * DIM))[lane];
    const float2 vb = ((const float2*)(img + row * DIM))[lane];

    const float ssa = wave_sum32(va.x * va.x + va.y * va.y);
    const float ssb = wave_sum32(vb.x * vb.x + vb.y * vb.y);
    const float sa  = 1.0f / fmaxf(sqrtf(ssa), EPS_N);   // torch cosine_similarity eps clamp
    const float sb  = 1.0f / fmaxf(sqrtf(ssb), EPS_N);

    v2h ha, hb;
    ha.x = (_Float16)(va.x * sa); ha.y = (_Float16)(va.y * sa);
    hb.x = (_Float16)(vb.x * sb); hb.y = (_Float16)(vb.y * sb);
    *(v2h*)(a_h + row * DIM + lane * 2) = ha;
    *(v2h*)(b_h + row * DIM + lane * 2) = hb;

    // diagonal of sim, from the f16-rounded values (consistent with the WMMA LSE)
    float p = (float)ha.x * (float)hb.x + (float)ha.y * (float)hb.y;
    p = wave_sum32(p);
    if (lane == 0) diag[row] = p * INV_T;
}

// ---------------- fused GEMM + exp row-sum pass ----------------
// out[m] += sum_n exp( (A[m]·B[n]) / T )   over this block's column chunk.
// Grid: 64*NCHUNK blocks of 256 threads (8 waves). Block b: rows 128*(b&63)..+127
// (one 16-row strip per wave), columns (b>>6)*16*TILES .. (TILES 16-wide tiles).
// B columns are bulk-staged in LDS, SUBT tiles at a time, so the inner loop has
// no barriers: pure ds_load_b128 -> v_wmma -> v_exp, pipelineable by the scheduler.
__global__ void lse_pass(const _Float16* __restrict__ A,
                         const _Float16* __restrict__ B,
                         float* __restrict__ out) {
    // SUBT*16 = 256 columns x 64 K halves, each column padded to 72 halves
    // (144B stride: l*36 mod 64 covers 16 distinct banks; keeps 16B alignment)
    __shared__ __align__(16) _Float16 tile[SUBT * 16 * 72];   // 36 KB

    const int tid  = threadIdx.x;
    const int wave = tid >> 5;
    const int lane = tid & 31;
    const int lmod = lane & 15;
    const int hsel = lane >> 4;            // which 16-lane half

    const int rowBlock = blockIdx.x & 63;
    const int chunk    = blockIdx.x >> 6;
    const int m0       = (rowBlock * 8 + wave) * 16;   // this wave's 16-row strip
    const int ntBase   = chunk * TILES;

    // A fragments (ISA 16-bit A 16x32 layout), invariant across the column loop.
    // lane l: row m0+(l%16); elems 0-7 = K c0..c0+7, elems 8-15 = K c0+16..c0+23,
    // c0 = (l>=16)? 8 : 0; second fragment is the K=32..63 half of the row.
    const _Float16* arow = A + (m0 + lmod) * DIM;
    const int c0 = hsel * 8;
    v8h a0lo = *(const v8h*)(arow + c0);
    v8h a0hi = *(const v8h*)(arow + c0 + 16);
    v8h a1lo = *(const v8h*)(arow + 32 + c0);
    v8h a1hi = *(const v8h*)(arow + 32 + c0 + 16);
    v16h afrag0 = __builtin_shufflevector(a0lo, a0hi, 0,1,2,3,4,5,6,7,8,9,10,11,12,13,14,15);
    v16h afrag1 = __builtin_shufflevector(a1lo, a1hi, 0,1,2,3,4,5,6,7,8,9,10,11,12,13,14,15);

    float acc[8];
#pragma unroll
    for (int r = 0; r < 8; ++r) acc[r] = 0.0f;

    const uint4* __restrict__ Bg4 = (const uint4*)B;
    unsigned* tilew = (unsigned*)tile;

    for (int st = 0; st < NSTAGE; ++st) {
        const int colBase = (ntBase + st * SUBT) * 16;   // first global column staged
        __syncthreads();
        // stage 256 cols x 32 dwords = 8192 dwords; 8 x uint4 (16B) per thread
#pragma unroll
        for (int j = 0; j < 8; ++j) {
            const int idx4 = (tid + j * 256) * 4;        // dword index, multiple of 4
            const int col  = idx4 >> 5;                  // 0..255
            const int dw   = idx4 & 31;                  // 0,4,..,28
            *(uint4*)(tilew + col * 36 + dw) = Bg4[((colBase + col) * 32 + dw) >> 2];
        }
        __syncthreads();

#pragma unroll 4
        for (int t = 0; t < SUBT; ++t) {
            // B fragments (ISA 16-bit B 32x16 layout): lane l = column l%16,
            // elems 0-7 = K c0..c0+7, elems 8-15 = K c0+16..c0+23, c0 = (l/16)*8 (+32 for the second fragment)
            const _Float16* bcol = tile + (t * 16 + lmod) * 72 + hsel * 8;
            v8h b0lo = *(const v8h*)(bcol);
            v8h b0hi = *(const v8h*)(bcol + 16);
            v8h b1lo = *(const v8h*)(bcol + 32);
            v8h b1hi = *(const v8h*)(bcol + 48);
            v16h bfrag0 = __builtin_shufflevector(b0lo, b0hi, 0,1,2,3,4,5,6,7,8,9,10,11,12,13,14,15);
            v16h bfrag1 = __builtin_shufflevector(b1lo, b1hi, 0,1,2,3,4,5,6,7,8,9,10,11,12,13,14,15);

            v8f c = {};
            c = __builtin_amdgcn_wmma_f32_16x16x32_f16(false, afrag0, false, bfrag0,
                                                       (short)0, c, false, false);
            c = __builtin_amdgcn_wmma_f32_16x16x32_f16(false, afrag1, false, bfrag1,
                                                       (short)0, c, false, false);

            // |arg| = |cos * 10 * log2e| <= 14.5 -> no under/overflow possible:
            // raw native exp2 (single v_exp_f32, no libm denorm guards).
#pragma unroll
            for (int r = 0; r < 8; ++r)
                acc[r] += __builtin_amdgcn_exp2f(c[r] * (INV_T * LOG2E));
        }
    }

    // C layout: VGPR r of lane l holds S[r + 8*(l/16)][l%16] -> reduce the 16
    // lanes of each half; lanes with l%16==0 carry rows m0+r and m0+8+r.
#pragma unroll
    for (int r = 0; r < 8; ++r) {
        const float v = half_sum16(acc[r]);
        if (lmod == 0) atomicAdd(&out[m0 + hsel * 8 + r], v);
    }
}

// ---------------- final scalar reduction ----------------
// loss = mean_i( log(rs[i]) + log(cs[i]) - 2*diag[i] ) / 2
__global__ void finalize_kernel(const float* __restrict__ rs,
                                const float* __restrict__ cs,
                                const float* __restrict__ dg,
                                float* __restrict__ out) {
    __shared__ float red[8];
    float s = 0.0f;
    for (int i = threadIdx.x; i < N_TOTAL; i += 256) {
        // rs/cs in [0.37, 1.8e8]: native v_log_f32 (log2) is accurate here
        const float l2 = __builtin_amdgcn_logf(rs[i]) + __builtin_amdgcn_logf(cs[i]);
        s += l2 * LN2 - 2.0f * dg[i];
    }
    s = wave_sum32(s);
    const int lane = threadIdx.x & 31;
    const int wave = threadIdx.x >> 5;
    if (lane == 0) red[wave] = s;
    __syncthreads();
    if (threadIdx.x == 0) {
        float tot = 0.0f;
#pragma unroll
        for (int w = 0; w < 8; ++w) tot += red[w];
        out[0] = tot / (2.0f * (float)N_TOTAL);
    }
}

extern "C" void kernel_launch(void* const* d_in, const int* in_sizes, int n_in,
                              void* d_out, int out_size, void* d_ws, size_t ws_size,
                              hipStream_t stream) {
    (void)in_sizes; (void)n_in; (void)out_size; (void)ws_size;
    const float* ts  = (const float*)d_in[0];
    const float* img = (const float*)d_in[1];
    float* out = (float*)d_out;

    // workspace layout (~2.1 MB total)
    char* ws = (char*)d_ws;
    _Float16* a_h    = (_Float16*)(ws);                    // 8192*64 f16 = 1 MB
    _Float16* b_h    = (_Float16*)(ws + 1048576);          // 1 MB
    float*    rowsum = (float*)(ws + 2097152);             // 32 KB
    float*    colsum = rowsum + N_TOTAL;                   // 32 KB (contiguous)
    float*    diag   = colsum + N_TOTAL;                   // 32 KB

    zero_kernel<<<(2 * N_TOTAL) / 256, 256, 0, stream>>>(rowsum, 2 * N_TOTAL);
    normalize_kernel<<<N_TOTAL / 8, 256, 0, stream>>>(ts, img, a_h, b_h, diag);
    lse_pass<<<64 * NCHUNK, 256, 0, stream>>>(a_h, b_h, rowsum);   // row LSE of S
    lse_pass<<<64 * NCHUNK, 256, 0, stream>>>(b_h, a_h, colsum);   // col LSE = row LSE of S^T
    finalize_kernel<<<1, 256, 0, stream>>>(rowsum, colsum, diag, out);
}
